// SparseConvEncoder_13666585936640
// MI455X (gfx1250) — matrix operand order repaired and checked by reading, not compiled, not run on hardware
//
#include <hip/hip_runtime.h>

// ---------------------------------------------------------------------------
// CDNA5 (gfx1250) implicit-GEMM 3D conv encoder, v_wmma_f32_16x16x32_bf16.
// bf16 end-to-end data path: weights pre-converted to bf16 (prep kernel),
// activations stored bf16 between layers (identical numerics to converting at
// the WMMA input, half the HBM traffic). fp32 only at the network input/output.
// Block = 8 wave32's covering BM x 8 output tiles of 16x16; per 32-deep K
// chunk the block stages A/B fragments into LDS in WMMA register layout and
// each wave reuses one A fragment against WT B fragments (WT WMMAs/chunk).
// ---------------------------------------------------------------------------

typedef __attribute__((ext_vector_type(16))) __bf16          v16bf;
typedef __attribute__((ext_vector_type(16))) unsigned short  v16us;
typedef __attribute__((ext_vector_type(8)))  float           v8f;
typedef unsigned short ushort_t;

// float -> bf16 round-to-nearest-even
__device__ __forceinline__ ushort_t f2bf(float f) {
  union { float f; unsigned u; } c; c.f = f;
  unsigned u = c.u;
  unsigned r = u + 0x7FFFu + ((u >> 16) & 1u);
  return (ushort_t)(r >> 16);
}

// typed helpers: load element as raw bf16 bits / store float as element type
__device__ __forceinline__ ushort_t load_bf(const float* p, int i)    { return f2bf(p[i]); }
__device__ __forceinline__ ushort_t load_bf(const ushort_t* p, int i) { return p[i]; }
__device__ __forceinline__ void store_el(float* p, int i, float v)    { p[i] = v; }
__device__ __forceinline__ void store_el(ushort_t* p, int i, float v) { p[i] = f2bf(v); }

// weight fp32 -> bf16 pre-conversion
__global__ __launch_bounds__(256) void wcvt(const float* __restrict__ s,
                                            ushort_t* __restrict__ d, int n) {
  const int i = blockIdx.x * 256 + threadIdx.x;
  if (i < n) d[i] = f2bf(s[i]);
}

// GEMM view: M = COUT, N = 2*DOUT^3, K = CIN*KS^3.  8 N-tiles per block.
template <int CIN, int COUT, int DIN, int DOUT, int KS, int STRIDE, int PAD,
          bool BN_EP, int BM, typename TIN, typename TOUT>
__global__ __launch_bounds__(256)
void conv_wmma(const TIN* __restrict__ x, const ushort_t* __restrict__ w,
               const float* __restrict__ gam, const float* __restrict__ bet,
               const float* __restrict__ mu,  const float* __restrict__ var,
               TOUT* __restrict__ out) {
  constexpr int BN      = 8;
  constexpr int WT      = BM * BN / 8;      // WMMAs per wave per chunk
  constexpr int KVOL    = KS * KS * KS;
  constexpr int K       = CIN * KVOL;
  constexpr int DV_IN   = DIN * DIN * DIN;
  constexpr int DV      = DOUT * DOUT * DOUT;
  constexpr int TILES_M = COUT / 16;
  constexpr int NCHUNK  = (K + 31) / 32;
  constexpr int GRID_M  = TILES_M / BM;
  constexpr int A_IT    = BM * 2;           // BM*512 elems / 256 threads
  constexpr int B_IT    = BN * 2;

  // Fragment-layout LDS: element (tile, fragLane, e) at [tile*512 + fragLane*16 + e]
  __shared__ __align__(32) ushort_t sA[BM * 512];
  __shared__ __align__(32) ushort_t sB[BN * 512];

  const int tid  = threadIdx.x;
  const int lane = tid & 31;
  const int wave = tid >> 5;
  const int wm   = wave % BM;
  const int wn0  = (wave / BM) * WT;
  const int m0   = (blockIdx.x % GRID_M) * BM * 16;
  const int n0   = (blockIdx.x / GRID_M) * BN * 16;

  // ---- k-invariant staging address math, hoisted out of the K loop ----
  int a_idx[A_IT], a_kl[A_IT];
#pragma unroll
  for (int it = 0; it < A_IT; ++it) {
    const int i = it * 256 + tid;
    const int mt = i >> 9, L = (i >> 4) & 31, e = i & 15;
    const int j = e >> 1, t = e & 1;
    a_kl[it]  = (j < 4 ? 2 * j : 16 + 2 * (j - 4)) + (L >> 4) * 8 + t;  // A frag layout
    a_idx[it] = (m0 + mt * 16 + (L & 15)) * K;
  }
  int b_off[B_IT], b_id0[B_IT], b_ih0[B_IT], b_iw0[B_IT];
#pragma unroll
  for (int it = 0; it < B_IT; ++it) {
    const int i = it * 256 + tid;
    const int nt = i >> 9, L = (i >> 4) & 31;
    const int n    = n0 + nt * 16 + (L & 15);
    const int bidx = n / DV;
    const int sp   = n - bidx * DV;
    const int od = sp / (DOUT * DOUT), oh = (sp / DOUT) % DOUT, ow = sp % DOUT;
    const int id0 = od * STRIDE - PAD, ih0 = oh * STRIDE - PAD, iw0 = ow * STRIDE - PAD;
    b_id0[it] = id0; b_ih0[it] = ih0; b_iw0[it] = iw0;
    b_off[it] = bidx * CIN * DV_IN + id0 * DIN * DIN + ih0 * DIN + iw0;
  }

  v8f acc[WT];
#pragma unroll
  for (int t = 0; t < WT; ++t) acc[t] = (v8f){};

  for (int kc = 0; kc < NCHUNK; ++kc) {
    const int k0 = kc * 32;

    // ---- stage bf16 weights -> sA (pure copy) ----
#pragma unroll
    for (int it = 0; it < A_IT; ++it) {
      const int k = k0 + a_kl[it];
      sA[it * 256 + tid] = (k < K) ? w[a_idx[it] + k] : (ushort_t)0;
    }
    // ---- stage zero-padded im2col patch -> sB ----
#pragma unroll
    for (int it = 0; it < B_IT; ++it) {
      const int i  = it * 256 + tid;
      const int kl = (((i >> 4) & 31) >> 4) * 8 + ((i & 15) < 8 ? (i & 15) : (i & 15) + 8);   // B frag layout
      const int k  = k0 + kl;
      ushort_t v = 0;
      if (k < K) {
        const int ci = k / KVOL;
        const int r  = k - ci * KVOL;
        const int kd = r / 9, kh = (r / 3) % 3, kw = r % 3;
        const int id = b_id0[it] + kd, ih = b_ih0[it] + kh, iw = b_iw0[it] + kw;
        if ((unsigned)id < (unsigned)DIN && (unsigned)ih < (unsigned)DIN &&
            (unsigned)iw < (unsigned)DIN)
          v = load_bf(x, b_off[it] + ci * DV_IN + kd * DIN * DIN + kh * DIN + kw);
      }
      sB[i] = v;
    }
    __syncthreads();

    // ---- one A fragment reused against WT B fragments ----
    const v16us au = *reinterpret_cast<const v16us*>(&sA[(wm * 32 + lane) * 16]);
    const v16bf A  = __builtin_bit_cast(v16bf, au);
#pragma unroll
    for (int t = 0; t < WT; ++t) {
      const v16us bu = *reinterpret_cast<const v16us*>(&sB[((wn0 + t) * 32 + lane) * 16]);
      acc[t] = __builtin_amdgcn_wmma_f32_16x16x32_bf16(
          false, A, false, __builtin_bit_cast(v16bf, bu),
          (short)0, acc[t], false, false);
    }
    __syncthreads();
  }

  // ---- epilogue: fused inference-BN + ReLU, scatter NCDHW ----
  // C/D layout: VGPR r -> M = (lane>>4)*8 + r ; N = lane & 15.
  const int half = lane >> 4, l15 = lane & 15;
  int obase[WT];
#pragma unroll
  for (int t = 0; t < WT; ++t) {
    const int n    = n0 + (wn0 + t) * 16 + l15;
    const int bidx = n / DV;
    const int sp   = n - bidx * DV;
    obase[t] = bidx * COUT * DV + sp;
  }
#pragma unroll
  for (int r = 0; r < 8; ++r) {
    const int co = m0 + wm * 16 + half * 8 + r;
    float inv = 1.f, sh = 0.f;
    if (BN_EP) {
      inv = gam[co] * rsqrtf(var[co] + 1e-5f);
      sh  = bet[co] - mu[co] * inv;
    }
#pragma unroll
    for (int t = 0; t < WT; ++t) {
      float val = acc[t][r];
      if (BN_EP) val = fmaxf(val * inv + sh, 0.f);
      store_el(out, obase[t] + co * DV, val);
    }
  }
}

extern "C" void kernel_launch(void* const* d_in, const int* in_sizes, int n_in,
                              void* d_out, int out_size, void* d_ws, size_t ws_size,
                              hipStream_t stream) {
  (void)in_sizes; (void)n_in; (void)out_size; (void)ws_size;

  const float* x = (const float*)d_in[0];
  const float* w1 = (const float*)d_in[1];
  const float* g1 = (const float*)d_in[2];
  const float* b1 = (const float*)d_in[3];
  const float* m1 = (const float*)d_in[4];
  const float* v1 = (const float*)d_in[5];
  const float* w2 = (const float*)d_in[6];
  const float* g2 = (const float*)d_in[7];
  const float* b2 = (const float*)d_in[8];
  const float* m2 = (const float*)d_in[9];
  const float* v2 = (const float*)d_in[10];
  const float* w3 = (const float*)d_in[11];
  const float* g3 = (const float*)d_in[12];
  const float* b3 = (const float*)d_in[13];
  const float* m3 = (const float*)d_in[14];
  const float* v3 = (const float*)d_in[15];
  const float* w4 = (const float*)d_in[16];
  const float* g4 = (const float*)d_in[17];
  const float* b4 = (const float*)d_in[18];
  const float* m4 = (const float*)d_in[19];
  const float* v4 = (const float*)d_in[20];
  const float* w5 = (const float*)d_in[21];
  const float* g5 = (const float*)d_in[22];
  const float* b5 = (const float*)d_in[23];
  const float* m5 = (const float*)d_in[24];
  const float* v5 = (const float*)d_in[25];
  const float* w_out = (const float*)d_in[26];
  float* out = (float*)d_out;

  // Workspace (all bf16/ushort), fully overwritten every call.
  ushort_t* h1  = (ushort_t*)d_ws;                 // 2*32*64^3  = 16,777,216
  ushort_t* h2  = h1 + (size_t)16777216;           // 2*64*32^3  =  4,194,304
  ushort_t* h3  = h2 + (size_t)4194304;            // 2*128*16^3 =  1,048,576
  ushort_t* h4  = h3 + (size_t)1048576;            // 2*256*8^3  =    262,144
  ushort_t* h5  = h4 + (size_t)262144;             // 2*512*4^3  =     65,536
  ushort_t* wb1 = h5 + (size_t)65536;              // 32*4*27    =      3,456
  ushort_t* wb2 = wb1 + (size_t)3456;              // 64*32*27   =     55,296
  ushort_t* wb3 = wb2 + (size_t)55296;             // 128*64*27  =    221,184
  ushort_t* wb4 = wb3 + (size_t)221184;            // 256*128*27 =    884,736
  ushort_t* wb5 = wb4 + (size_t)884736;            // 512*256*27 =  3,538,944
  ushort_t* wbo = wb5 + (size_t)3538944;           // 256*512    =    131,072

  // ---- weight fp32 -> bf16 pre-conversion ----
  wcvt<<<(3456 + 255) / 256, 256, 0, stream>>>(w1, wb1, 3456);
  wcvt<<<(55296 + 255) / 256, 256, 0, stream>>>(w2, wb2, 55296);
  wcvt<<<(221184 + 255) / 256, 256, 0, stream>>>(w3, wb3, 221184);
  wcvt<<<(884736 + 255) / 256, 256, 0, stream>>>(w4, wb4, 884736);
  wcvt<<<(3538944 + 255) / 256, 256, 0, stream>>>(w5, wb5, 3538944);
  wcvt<<<(131072 + 255) / 256, 256, 0, stream>>>(w_out, wbo, 131072);

  const dim3 blk(256);  // 8 waves; tile grids divide exactly -> EXEC all-ones

  // grid = (TILES_M/BM) * (TILES_N/8)
  // L1: fp32 in -> bf16 out
  conv_wmma<4, 32, 128, 64, 3, 2, 1, true, 2, float, ushort_t>
      <<<4096, blk, 0, stream>>>(x, wb1, g1, b1, m1, v1, h1);
  // L2..L5: bf16 -> bf16
  conv_wmma<32, 64, 64, 32, 3, 2, 1, true, 4, ushort_t, ushort_t>
      <<<512, blk, 0, stream>>>(h1, wb2, g2, b2, m2, v2, h2);
  conv_wmma<64, 128, 32, 16, 3, 2, 1, true, 8, ushort_t, ushort_t>
      <<<64, blk, 0, stream>>>(h2, wb3, g3, b3, m3, v3, h3);
  conv_wmma<128, 256, 16, 8, 3, 2, 1, true, 8, ushort_t, ushort_t>
      <<<16, blk, 0, stream>>>(h3, wb4, g4, b4, m4, v4, h4);
  conv_wmma<256, 512, 8, 4, 3, 2, 1, true, 8, ushort_t, ushort_t>
      <<<4, blk, 0, stream>>>(h4, wb5, g5, b5, m5, v5, h5);
  // Output 1x1x1: bf16 in -> fp32 out, no BN/ReLU
  conv_wmma<512, 256, 4, 4, 1, 1, 0, false, 8, ushort_t, float>
      <<<2, blk, 0, stream>>>(h5, wbo, nullptr, nullptr, nullptr, nullptr, out);
}
